// VectorQuantizer_83734682403296
// MI455X (gfx1250) — matrix operand order repaired and checked by reading, not compiled, not run on hardware
//
#include <hip/hip_runtime.h>
#include <hip/hip_bf16.h>

// CDNA5 / gfx1250 VQ-VAE vector quantizer.
// Heavy op: 32768x8192 score matrix over D=256 via v_wmma_f32_16x16x32_f16,
// argmin fused in the epilogue. emb scaled by 8192 so f16 keeps precision.
// v2: 2 token-strips per wave (B reused 2x per LDS read), load-early /
//     ds_store-late double buffering so global latency hides behind WMMAs.

typedef __attribute__((ext_vector_type(16))) _Float16     v16h;
typedef __attribute__((ext_vector_type(8)))  float        v8f;
typedef __attribute__((ext_vector_type(8)))  unsigned int v8u;

#define N_TOK   32768          // 8 * 4096 tokens
#define K_EMB   8192           // codebook size
#define D_DIM   256            // embedding dim (GEMM K)
#define T_DIM   4096
#define KT_CNT  (K_EMB / 16)   // 512 column tiles
#define DB_CNT  (D_DIM / 32)   // 8 depth blocks per 16x16x32 WMMA
#define SCALE   8192.0f

// workspace layout (bytes)
#define A_OFF    0u            // 32768*256 halves = 16 MB (A fragments)
#define B_OFF    16777216u     // 8192*256 halves  =  4 MB (B fragments)
#define NORM_OFF 20971520u     // 8192 f32 norms
#define IDX_OFF  21004288u     // 32768 i32 indices
#define PART_OFF 21135360u     // 2048 f32 loss partials

// ---------------------------------------------------------------------------
// A-fragment layout (CDNA5 ISA, 16-bit A 16x32):
//   lane<16: M=lane,    VGPR v(0..3): K=2v,2v+1     ; v(4..7): K=+16
//   lane>=16: M=lane-16, same with K offset +8
// packed as uint32 pairs: aBuf[((s*8+db)*32+lane)*8 + v]
__global__ void vq_prep_a(const float* __restrict__ z, unsigned* __restrict__ aBuf) {
    int tid = threadIdx.x;
    int t   = blockIdx.x * 256 + tid;   // coalesced along T
    int b   = blockIdx.y;
    int p   = blockIdx.z;               // d-pair index 0..127
    int d   = p * 2;
    float z0 = z[((size_t)b * D_DIM + d)     * T_DIM + t];
    float z1 = z[((size_t)b * D_DIM + d + 1) * T_DIM + t];
    union { _Float16 h[2]; unsigned u; } pk;
    pk.h[0] = (_Float16)z0;
    pk.h[1] = (_Float16)z1;
    int n    = b * T_DIM + t;
    int s    = n >> 4;                  // 16-row strip
    int db   = p >> 4;                  // depth block (32 channels)
    int kp   = p & 15;                  // K-pair within block
    int lane = (n & 15) + (((kp >> 2) & 1) << 4);
    int v    = (kp & 3) | ((kp >> 3) << 2);
    aBuf[(((size_t)s * DB_CNT + db) * 32 + lane) * 8 + v] = pk.u;
}

// B-fragment layout (16-bit B 32x16, same K order as A):
//   lane<16: N=lane, VGPR v(0..3): K=2v,2v+1; v(4..7): K=+16; lane>=16: K offset +8
__global__ void vq_prep_b(const float* __restrict__ emb, unsigned* __restrict__ bBuf) {
    int p   = threadIdx.x;              // d-pair 0..127
    int col = blockIdx.x;               // codebook row 0..8191
    float2 e = ((const float2*)(emb + (size_t)col * D_DIM))[p];
    union { _Float16 h[2]; unsigned u; } pk;
    pk.h[0] = (_Float16)(e.x * SCALE);
    pk.h[1] = (_Float16)(e.y * SCALE);
    int db   = p >> 4;
    int q    = p & 15;
    int lane = (col & 15) + (((q >> 2) & 1) << 4);
    int v    = (q & 3) | ((q >> 3) << 2);
    int kt   = col >> 4;
    bBuf[(((size_t)kt * DB_CNT + db) * 32 + lane) * 8 + v] = pk.u;
}

__global__ void vq_norm(const float* __restrict__ emb, float* __restrict__ norms) {
    __shared__ float red[256];
    int row = blockIdx.x, tid = threadIdx.x;
    float v = emb[(size_t)row * D_DIM + tid] * SCALE;
    red[tid] = v * v;
    __syncthreads();
    for (int st = 128; st > 0; st >>= 1) {
        if (tid < st) red[tid] += red[tid + st];
        __syncthreads();
    }
    if (tid == 0) norms[row] = red[0];
}

// ---------------------------------------------------------------------------
// Hot kernel: 8 waves/workgroup; each wave owns TWO 16-token strips so every
// B fragment pulled from LDS feeds two WMMAs. emb tiles double-buffered in
// LDS with global loads issued before compute, ds_store after compute.
__global__ __launch_bounds__(256, 1) void vq_argmin(
        const unsigned* __restrict__ aBuf, const unsigned* __restrict__ bBuf,
        const float* __restrict__ norms, int* __restrict__ idxWs,
        int* __restrict__ idxOut) {
    __shared__ unsigned sB[2][2048];    // 2 x 8KB: one k-tile of B (8 db frags)
    __shared__ float    sNorm[K_EMB];   // 32KB: all scaled norms

    int tid  = threadIdx.x;
    int wave = tid >> 5;
    int lane = tid & 31;

    for (int i = tid; i < K_EMB; i += 256) sNorm[i] = norms[i];

    int s0 = (blockIdx.x * 8 + wave) * 2;   // two consecutive strips per wave
    int s1 = s0 + 1;
    v16h a0[DB_CNT], a1[DB_CNT];
#pragma unroll
    for (int db = 0; db < DB_CNT; ++db) {
        v8u r0 = ((const v8u*)aBuf)[((size_t)s0 * DB_CNT + db) * 32 + lane];
        v8u r1 = ((const v8u*)aBuf)[((size_t)s1 * DB_CNT + db) * 32 + lane];
        a0[db] = __builtin_bit_cast(v16h, r0);
        a1[db] = __builtin_bit_cast(v16h, r1);
    }

    float best0[8], best1[8];
    int   bestk0[8], bestk1[8];
#pragma unroll
    for (int r = 0; r < 8; ++r) {
        best0[r] = 3.4e38f; bestk0[r] = 0;
        best1[r] = 3.4e38f; bestk1[r] = 0;
    }

    { // stage kt = 0
        const uint4* g = (const uint4*)bBuf;
        uint4* sh = (uint4*)sB[0];
        sh[tid]       = g[tid];
        sh[tid + 256] = g[tid + 256];
    }
    __syncthreads();

    for (int kt = 0; kt < KT_CNT; ++kt) {
        int  cur     = kt & 1;
        bool hasNext = (kt + 1 < KT_CNT);

        // issue next tile's global loads now; latency covered by 16 WMMAs
        uint4 stg0, stg1;
        if (hasNext) {
            const uint4* g = (const uint4*)(bBuf + (size_t)(kt + 1) * 2048);
            stg0 = g[tid];
            stg1 = g[tid + 256];
        }

        v8f acc0 = {0.f, 0.f, 0.f, 0.f, 0.f, 0.f, 0.f, 0.f};
        v8f acc1 = {0.f, 0.f, 0.f, 0.f, 0.f, 0.f, 0.f, 0.f};
#pragma unroll
        for (int db = 0; db < DB_CNT; ++db) {
            v8u raw = ((const v8u*)sB[cur])[db * 32 + lane];
            v16h bf = __builtin_bit_cast(v16h, raw);
            acc0 = __builtin_amdgcn_wmma_f32_16x16x32_f16(
                       false, a0[db], false, bf, (short)0, acc0, false, false);
            acc1 = __builtin_amdgcn_wmma_f32_16x16x32_f16(
                       false, a1[db], false, bf, (short)0, acc1, false, false);
        }

        // epilogue: score = ||e'||^2 - 2*SCALE*dot  (argmin-invariant scaling)
        int   kl = lane & 15;
        float kn = sNorm[kt * 16 + kl];
        int   k  = kt * 16 + kl;
#pragma unroll
        for (int r = 0; r < 8; ++r) {
            float c0 = kn - (2.0f * SCALE) * acc0[r];
            float c1 = kn - (2.0f * SCALE) * acc1[r];
            if (c0 < best0[r]) { best0[r] = c0; bestk0[r] = k; }
            if (c1 < best1[r]) { best1[r] = c1; bestk1[r] = k; }
        }

        // now commit the staged tile to the other LDS buffer
        if (hasNext) {
            uint4* sh = (uint4*)sB[cur ^ 1];
            sh[tid]       = stg0;
            sh[tid + 256] = stg1;
        }
        __syncthreads();
    }

    // reduce over the 16 lanes holding the same output row; ties -> smaller k
#pragma unroll
    for (int r = 0; r < 8; ++r) {
        float bv0 = best0[r];  int bk0 = bestk0[r];
        float bv1 = best1[r];  int bk1 = bestk1[r];
        for (int m = 1; m < 16; m <<= 1) {
            float ov0 = __shfl_xor(bv0, m, 32);
            int   ok0 = __shfl_xor(bk0, m, 32);
            float ov1 = __shfl_xor(bv1, m, 32);
            int   ok1 = __shfl_xor(bk1, m, 32);
            if (ov0 < bv0 || (ov0 == bv0 && ok0 < bk0)) { bv0 = ov0; bk0 = ok0; }
            if (ov1 < bv1 || (ov1 == bv1 && ok1 < bk1)) { bv1 = ov1; bk1 = ok1; }
        }
        if ((lane & 15) == 0) {
            int half = (lane >> 4) << 3;    // C layout: lanes>=16 hold M=r+8
            int row0 = s0 * 16 + r + half;
            int row1 = s1 * 16 + r + half;
            idxWs[row0]  = bk0;  idxOut[row0] = bk0;
            idxWs[row1]  = bk1;  idxOut[row1] = bk1;
        }
    }
}

// ---------------------------------------------------------------------------
// Gather emb[idx] -> [B,D,T] via LDS transpose tile; fused loss partials.
__global__ void vq_gather(const float* __restrict__ z, const float* __restrict__ emb,
                          const int* __restrict__ idxWs, float* __restrict__ zq,
                          float* __restrict__ partial) {
    __shared__ float tile[16 * 257];    // pad avoids bank conflicts
    __shared__ float red[256];
    int tid = threadIdx.x;
    int n0  = blockIdx.x * 16;
    int b   = n0 >> 12;
    int t0  = n0 & 4095;
#pragma unroll
    for (int i = 0; i < 16; ++i) {
        int row = idxWs[n0 + i];
        tile[i * 257 + tid] = emb[(size_t)row * D_DIM + tid];   // coalesced
    }
    __syncthreads();
    float lsum = 0.f;
    int tl = tid & 15;
    int dh = tid >> 4;
#pragma unroll
    for (int j = 0; j < 16; ++j) {
        int d = j * 16 + dh;
        float q = tile[tl * 257 + d];
        size_t off = ((size_t)b * D_DIM + d) * T_DIM + t0 + tl; // 64B chunks
        float diff = q - z[off];
        zq[off] = q;
        lsum += diff * diff;
    }
    red[tid] = lsum;
    __syncthreads();
    for (int st = 128; st > 0; st >>= 1) {
        if (tid < st) red[tid] += red[tid + st];
        __syncthreads();
    }
    if (tid == 0) partial[blockIdx.x] = red[0];
}

__global__ void vq_loss(const float* __restrict__ partial, float* __restrict__ lossOut) {
    __shared__ float red[256];
    int tid = threadIdx.x;
    float s = 0.f;
    for (int i = tid; i < 2048; i += 256) s += partial[i];
    red[tid] = s;
    __syncthreads();
    for (int st = 128; st > 0; st >>= 1) {
        if (tid < st) red[tid] += red[tid + st];
        __syncthreads();
    }
    // forward loss = (1 + BETA) * mean((z_q - z)^2)
    if (tid == 0) lossOut[0] = red[0] * (1.25f / 8388608.0f);
}

// ---------------------------------------------------------------------------
extern "C" void kernel_launch(void* const* d_in, const int* in_sizes, int n_in,
                              void* d_out, int out_size, void* d_ws, size_t ws_size,
                              hipStream_t stream) {
    const float* z   = (const float*)d_in[0];   // [8,256,4096] f32
    const float* emb = (const float*)d_in[1];   // [8192,256]  f32

    float* outF    = (float*)d_out;
    float* zq      = outF;                      // 8,388,608 floats
    float* lossOut = outF + 8388608;            // 1 float
    int*   idxOut  = (int*)(outF + 8388609);    // 32768 ints (bit-pattern region)

    char* ws = (char*)d_ws;
    unsigned* aBuf   = (unsigned*)(ws + A_OFF);
    unsigned* bBuf   = (unsigned*)(ws + B_OFF);
    float*    norms  = (float*)(ws + NORM_OFF);
    int*      idxWs  = (int*)(ws + IDX_OFF);
    float*    part   = (float*)(ws + PART_OFF);

    vq_prep_a<<<dim3(16, 8, 128), 256, 0, stream>>>(z, aBuf);
    vq_prep_b<<<dim3(8192), 128, 0, stream>>>(emb, bBuf);
    vq_norm  <<<dim3(8192), 256, 0, stream>>>(emb, norms);
    vq_argmin<<<dim3(128),  256, 0, stream>>>(aBuf, bBuf, norms, idxWs, idxOut);
    vq_gather<<<dim3(2048), 256, 0, stream>>>(z, emb, idxWs, zq, part);
    vq_loss  <<<dim3(1),    256, 0, stream>>>(part, lossOut);
}
